// LightNetworkEnc_77206332113749
// MI455X (gfx1250) — matrix operand order repaired and checked by reading, not compiled, not run on hardware
//
#include <hip/hip_runtime.h>
#include <hip/hip_bf16.h>
#include <math.h>

typedef __bf16 bf16_t;
typedef __attribute__((ext_vector_type(16))) __bf16 v16bf;
typedef __attribute__((ext_vector_type(8)))  float  v8f;

#define N_NODES 100000
#define N_EDGES 800000
#define N_TILES 6250            // N_NODES / 16 (exact)
#define WPB 8                   // waves per block
#define BLOCK_THREADS 256

__device__ static inline bf16_t f2b(float f) { return (bf16_t)f; }

__device__ static inline v8f wmma_bf16(v16bf a, v16bf b, v8f c) {
  return __builtin_amdgcn_wmma_f32_16x16x32_bf16(false, a, false, b, (short)0, c, false, false);
}

// ---------------- fragment loaders (layouts per CDNA5 ISA) --------

// A-matrix 16x32 bf16 from LDS row-major (stride ld), K window [k0, k0+32)
// lanes 0-15: K kb..kb+7 and kb+16..kb+23 -> two contiguous 16B runs (ds_load_b128)
__device__ static inline v16bf load_a_lds(const bf16_t* base, int ld, int lane, int k0) {
  int m  = lane & 15;
  int kb = k0 + ((lane >> 4) << 3);
  const bf16_t* p = base + m * ld + kb;
  v16bf a;
#pragma unroll
  for (int e = 0; e < 8; ++e) a[e] = p[e];
#pragma unroll
  for (int e = 0; e < 8; ++e) a[8 + e] = p[16 + e];
  return a;
}

// A-matrix 16x32 bf16 from global, rows padded/zero-filled to ldk (mult of 32)
__device__ static inline v16bf load_a_global(const bf16_t* x, int row0, int ldk, int lane, int k0) {
  int m  = lane & 15;
  int kb = k0 + ((lane >> 4) << 3);
  const bf16_t* p = x + (size_t)(row0 + m) * ldk + kb;
  v16bf a;
#pragma unroll
  for (int e = 0; e < 8; ++e) a[e] = p[e];
#pragma unroll
  for (int e = 0; e < 8; ++e) a[8 + e] = p[16 + e];
  return a;
}

// B fragment from swizzled LDS: layout [frag][lane][16 bf16 contiguous]
// frag base must point at a 512-element (1KB) fragment block
__device__ static inline v16bf load_b_frag(const bf16_t* fragBase, int lane) {
  const bf16_t* p = fragBase + (lane << 4);
  v16bf b;
#pragma unroll
  for (int e = 0; e < 16; ++e) b[e] = p[e];
  return b;
}

// Stage weight matrix w[di][dout] (fp32) into swizzled bf16 LDS:
// dst layout: [ct][ks][lane][16], frag block = ((ct*nks)+ks)*512
// K zero-padded to Kp (multiple of 32); dout must be a multiple of 16.
__device__ static inline void stage_w_swz(const float* __restrict__ w, bf16_t* dst,
                                          int Kp, int di, int dout, int tid) {
  int nks = Kp >> 5;
  for (int idx = tid; idx < Kp * dout; idx += BLOCK_THREADS) {
    int k = idx / dout, n = idx - k * dout;
    float v = (k < di) ? w[k * dout + n] : 0.f;
    int ct = n >> 4, nn = n & 15;
    int ks = k >> 5, kk = k & 31;
    int l = nn + ((kk & 8) << 1);
    int e = (kk & 7) | ((kk & 16) >> 1);
    dst[(((ct * nks + ks) * 32 + l) << 4) + e] = f2b(v);
  }
}

// ===================== encoder: concat + 64->32->64->128 ======================
__global__ void __launch_bounds__(BLOCK_THREADS) enc_kernel(
    const float* __restrict__ one_hot, const float* __restrict__ feat,
    const float* __restrict__ w0, const float* __restrict__ b0,
    const float* __restrict__ w1, const float* __restrict__ b1,
    const float* __restrict__ w2, const float* __restrict__ b2,
    bf16_t* __restrict__ xout)   // [N][128] bf16
{
  __shared__ bf16_t sW0[64 * 32];    // swizzled: nct=2, nks=2
  __shared__ bf16_t sW1[32 * 64];    // swizzled: nct=4, nks=1
  __shared__ bf16_t sW2[64 * 128];   // swizzled: nct=8, nks=2
  __shared__ float  sB0[32], sB1[64], sB2[128];
  __shared__ bf16_t sA[WPB][16 * 64];
  __shared__ bf16_t sB[WPB][16 * 64];

  int tid = threadIdx.x;
  stage_w_swz(w0, sW0, 64, 64, 32, tid);
  stage_w_swz(w1, sW1, 32, 32, 64, tid);
  stage_w_swz(w2, sW2, 64, 64, 128, tid);
  if (tid < 32)  sB0[tid] = b0[tid];
  if (tid < 64)  sB1[tid] = b1[tid];
  if (tid < 128) sB2[tid] = b2[tid];
  __syncthreads();

  int wave = tid >> 5, lane = tid & 31;
  int tile = blockIdx.x * WPB + wave;
  if (tile >= N_TILES) return;
  int row0 = tile * 16;
  bf16_t* A = sA[wave];
  bf16_t* B = sB[wave];

  // stage concat(one_hot[20], features[44]) -> 16x64 bf16
  for (int i = lane; i < 16 * 64; i += 32) {
    int r = i >> 6, c = i & 63;
    float v = (c < 20) ? one_hot[(size_t)(row0 + r) * 20 + c]
                       : feat[(size_t)(row0 + r) * 44 + (c - 20)];
    A[i] = f2b(v);
  }
  int n = lane & 15, moff = (lane >> 4) * 8;

  // layer 0: 64 -> 32 (relu) into B (ld 32)
#pragma unroll
  for (int nt = 0; nt < 2; ++nt) {
    v8f acc = {0.f, 0.f, 0.f, 0.f, 0.f, 0.f, 0.f, 0.f};
#pragma unroll
    for (int ks = 0; ks < 2; ++ks)
      acc = wmma_bf16(load_a_lds(A, 64, lane, ks * 32),
                      load_b_frag(sW0 + ((nt * 2 + ks) << 9), lane), acc);
    float bb = sB0[nt * 16 + n];
#pragma unroll
    for (int v = 0; v < 8; ++v)
      B[(moff + v) * 32 + nt * 16 + n] = f2b(fmaxf(acc[v] + bb, 0.f));
  }
  // layer 1: 32 -> 64 (relu) into A (ld 64)
#pragma unroll
  for (int nt = 0; nt < 4; ++nt) {
    v8f acc = {0.f, 0.f, 0.f, 0.f, 0.f, 0.f, 0.f, 0.f};
    acc = wmma_bf16(load_a_lds(B, 32, lane, 0), load_b_frag(sW1 + (nt << 9), lane), acc);
    float bb = sB1[nt * 16 + n];
#pragma unroll
    for (int v = 0; v < 8; ++v)
      A[(moff + v) * 64 + nt * 16 + n] = f2b(fmaxf(acc[v] + bb, 0.f));
  }
  // layer 2: 64 -> 128 (relu) straight to global bf16 [N][128]
#pragma unroll
  for (int nt = 0; nt < 8; ++nt) {
    v8f acc = {0.f, 0.f, 0.f, 0.f, 0.f, 0.f, 0.f, 0.f};
#pragma unroll
    for (int ks = 0; ks < 2; ++ks)
      acc = wmma_bf16(load_a_lds(A, 64, lane, ks * 32),
                      load_b_frag(sW2 + ((nt * 2 + ks) << 9), lane), acc);
    float bb = sB2[nt * 16 + n];
#pragma unroll
    for (int v = 0; v < 8; ++v)
      xout[(size_t)(row0 + moff + v) * 128 + nt * 16 + n] = f2b(fmaxf(acc[v] + bb, 0.f));
  }
}

// ========== conv GEMM: C[N, 5*do] = x @ [Ws | Wr0 Wr1 Wr2 Wr3] ===============
// cols [0,do): agg = acc + bias (fp32);  cols [do,5do): h (bf16, layout [N][4][do])
// xin rows padded to ldk = KSTEPS*32 with zeros.
template <int KSTEPS>
__global__ void __launch_bounds__(BLOCK_THREADS) conv_gemm_kernel(
    const bf16_t* __restrict__ xin,  // [N][ldk] bf16 (zero-padded)
    const float* __restrict__ wr,    // [4][di][do]
    const float* __restrict__ wsw,   // [di][do]
    const float* __restrict__ bias,  // [do]
    float* __restrict__ agg,         // [N][do]
    bf16_t* __restrict__ hbuf,       // [N][4*do]
    int di, int dout)
{
  __shared__ bf16_t sBt[KSTEPS * 512];   // swizzled B tile: [ks][lane][16]

  int tid = threadIdx.x, wave = tid >> 5, lane = tid & 31;
  int tile = blockIdx.x * WPB + wave;
  bool valid = (tile < N_TILES);
  int row0 = tile * 16;
  const int ldk = KSTEPS * 32;

  v16bf afrag[KSTEPS];
  if (valid) {
#pragma unroll
    for (int ks = 0; ks < KSTEPS; ++ks)
      afrag[ks] = load_a_global(xin, row0, ldk, lane, ks * 32);
  }

  const int totalCols = 5 * dout;
  const int nColTiles = (totalCols + 15) >> 4;

  for (int ct = 0; ct < nColTiles; ++ct) {
    __syncthreads();
    // stage swizzled B tile (zero-padded in K and cols)
    for (int i = tid; i < KSTEPS * 512; i += BLOCK_THREADS) {
      int k = i >> 4, nn = i & 15;
      int c = ct * 16 + nn;
      float v = 0.f;
      if (k < di && c < totalCols) {
        if (c < dout) {
          v = wsw[(size_t)k * dout + c];
        } else {
          int cr = c - dout;
          int r = cr / dout, cc = cr - r * dout;
          v = wr[((size_t)r * di + k) * dout + cc];
        }
      }
      int ks = k >> 5, kk = k & 31;
      int l = nn + ((kk & 8) << 1);
      int e = (kk & 7) | ((kk & 16) >> 1);
      sBt[((ks * 32 + l) << 4) + e] = f2b(v);
    }
    __syncthreads();
    if (!valid) continue;

    v8f acc = {0.f, 0.f, 0.f, 0.f, 0.f, 0.f, 0.f, 0.f};
#pragma unroll
    for (int ks = 0; ks < KSTEPS; ++ks)
      acc = wmma_bf16(afrag[ks], load_b_frag(sBt + (ks << 9), lane), acc);

    int n = lane & 15, moff = (lane >> 4) * 8;
    int c = ct * 16 + n;
    if (c < dout) {
      float bb = bias[c];
#pragma unroll
      for (int v = 0; v < 8; ++v)
        agg[(size_t)(row0 + moff + v) * dout + c] = acc[v] + bb;
    } else if (c < totalCols) {
      int cr = c - dout;
      int r = cr / dout, cc = cr - r * dout;
#pragma unroll
      for (int v = 0; v < 8; ++v)
        hbuf[((size_t)(row0 + moff + v) * 4 + r) * dout + cc] = f2b(acc[v]);
    }
  }
}

// ====================== edge scatter: agg[dst] += h[src][etype] ==============
__global__ void __launch_bounds__(BLOCK_THREADS) scatter_kernel(
    const bf16_t* __restrict__ hbuf, const int* __restrict__ esrc,
    const int* __restrict__ edst, const int* __restrict__ etyp,
    float* __restrict__ agg, int nEdges, int dout)
{
  int wave = (int)((blockIdx.x * (unsigned)blockDim.x + threadIdx.x) >> 5);
  int lane = threadIdx.x & 31;
  int nWaves = (int)((gridDim.x * (unsigned)blockDim.x) >> 5);
  for (int e = wave; e < nEdges; e += nWaves) {
    int s = esrc[e], d = edst[e], r = etyp[e];
    const bf16_t* hrow = hbuf + ((size_t)s * 4 + r) * dout;
    float* arow = agg + (size_t)d * dout;
    for (int c = lane; c < dout; c += 32)
      atomicAdd(&arow[c], (float)hrow[c]);
  }
}

// ============== relu + convert to bf16, zero-padded rows to kpad =============
__global__ void __launch_bounds__(BLOCK_THREADS) relu_pad_bf16_kernel(
    const float* __restrict__ agg, bf16_t* __restrict__ xout,
    int dout, int kpad, int n /* = N * kpad */)
{
  int i = blockIdx.x * BLOCK_THREADS + threadIdx.x;
  if (i >= n) return;
  int row = i / kpad, c = i - row * kpad;
  float v = (c < dout) ? fmaxf(agg[(size_t)row * dout + c], 0.f) : 0.f;
  xout[i] = f2b(v);
}

// ============ SC head: 8->16->32->64->32->16 (relu) + 16->1 sigmoid ==========
// xin: [N][32] bf16, cols 8..31 pre-zeroed
__global__ void __launch_bounds__(BLOCK_THREADS) sc_kernel(
    const bf16_t* __restrict__ xin,
    const float* __restrict__ w0, const float* __restrict__ b0,
    const float* __restrict__ w1, const float* __restrict__ b1,
    const float* __restrict__ w2, const float* __restrict__ b2,
    const float* __restrict__ w3, const float* __restrict__ b3,
    const float* __restrict__ w4, const float* __restrict__ b4,
    const float* __restrict__ fw, const float* __restrict__ fb,
    float* __restrict__ out)
{
  __shared__ bf16_t sW0[32 * 16];   // swz nct=1 nks=1 (K 8 -> 32)
  __shared__ bf16_t sW1[32 * 32];   // swz nct=2 nks=1 (K 16 -> 32)
  __shared__ bf16_t sW2[32 * 64];   // swz nct=4 nks=1
  __shared__ bf16_t sW3[64 * 32];   // swz nct=2 nks=2
  __shared__ bf16_t sW4[32 * 16];   // swz nct=1 nks=1
  __shared__ float  sBi[16 + 32 + 64 + 32 + 16];
  __shared__ float  sFw[16];
  __shared__ float  sFb;
  __shared__ bf16_t sA[WPB][16 * 64];
  __shared__ bf16_t sB[WPB][16 * 64];

  int tid = threadIdx.x;
  stage_w_swz(w0, sW0, 32, 8, 16, tid);
  stage_w_swz(w1, sW1, 32, 16, 32, tid);
  stage_w_swz(w2, sW2, 32, 32, 64, tid);
  stage_w_swz(w3, sW3, 64, 64, 32, tid);
  stage_w_swz(w4, sW4, 32, 32, 16, tid);
  if (tid < 16) sBi[tid] = b0[tid];
  if (tid < 32) sBi[16 + tid] = b1[tid];
  if (tid < 64) sBi[48 + tid] = b2[tid];
  if (tid < 32) sBi[112 + tid] = b3[tid];
  if (tid < 16) sBi[144 + tid] = b4[tid];
  if (tid < 16) sFw[tid] = fw[tid];
  if (tid == 0) sFb = fb[0];
  __syncthreads();

  int wave = tid >> 5, lane = tid & 31;
  int tile = blockIdx.x * WPB + wave;
  if (tile >= N_TILES) return;
  int row0 = tile * 16;
  bf16_t* A = sA[wave];
  bf16_t* B = sB[wave];

  // stage 16x32 pre-padded input (contiguous copy)
  for (int i = lane; i < 16 * 32; i += 32)
    A[i] = xin[(size_t)row0 * 32 + i];
  int n = lane & 15, moff = (lane >> 4) * 8;

  // layer0: 8->16 -> B (ld 32, cols 16..31 zeroed for next Kpad=32)
  {
    v8f acc = {0.f, 0.f, 0.f, 0.f, 0.f, 0.f, 0.f, 0.f};
    acc = wmma_bf16(load_a_lds(A, 32, lane, 0), load_b_frag(sW0, lane), acc);
    float bb = sBi[n];
#pragma unroll
    for (int v = 0; v < 8; ++v) {
      B[(moff + v) * 32 + n]      = f2b(fmaxf(acc[v] + bb, 0.f));
      B[(moff + v) * 32 + 16 + n] = f2b(0.f);
    }
  }
  // layer1: 16->32 -> A (ld 32)
#pragma unroll
  for (int nt = 0; nt < 2; ++nt) {
    v8f acc = {0.f, 0.f, 0.f, 0.f, 0.f, 0.f, 0.f, 0.f};
    acc = wmma_bf16(load_a_lds(B, 32, lane, 0), load_b_frag(sW1 + (nt << 9), lane), acc);
    float bb = sBi[16 + nt * 16 + n];
#pragma unroll
    for (int v = 0; v < 8; ++v)
      A[(moff + v) * 32 + nt * 16 + n] = f2b(fmaxf(acc[v] + bb, 0.f));
  }
  // layer2: 32->64 -> B (ld 64)
#pragma unroll
  for (int nt = 0; nt < 4; ++nt) {
    v8f acc = {0.f, 0.f, 0.f, 0.f, 0.f, 0.f, 0.f, 0.f};
    acc = wmma_bf16(load_a_lds(A, 32, lane, 0), load_b_frag(sW2 + (nt << 9), lane), acc);
    float bb = sBi[48 + nt * 16 + n];
#pragma unroll
    for (int v = 0; v < 8; ++v)
      B[(moff + v) * 64 + nt * 16 + n] = f2b(fmaxf(acc[v] + bb, 0.f));
  }
  // layer3: 64->32 -> A (ld 32), K=64 in two steps
#pragma unroll
  for (int nt = 0; nt < 2; ++nt) {
    v8f acc = {0.f, 0.f, 0.f, 0.f, 0.f, 0.f, 0.f, 0.f};
#pragma unroll
    for (int ks = 0; ks < 2; ++ks)
      acc = wmma_bf16(load_a_lds(B, 64, lane, ks * 32),
                      load_b_frag(sW3 + ((nt * 2 + ks) << 9), lane), acc);
    float bb = sBi[112 + nt * 16 + n];
#pragma unroll
    for (int v = 0; v < 8; ++v)
      A[(moff + v) * 32 + nt * 16 + n] = f2b(fmaxf(acc[v] + bb, 0.f));
  }
  // layer4: 32->16 -> B (ld 16)
  {
    v8f acc = {0.f, 0.f, 0.f, 0.f, 0.f, 0.f, 0.f, 0.f};
    acc = wmma_bf16(load_a_lds(A, 32, lane, 0), load_b_frag(sW4, lane), acc);
    float bb = sBi[144 + n];
#pragma unroll
    for (int v = 0; v < 8; ++v)
      B[(moff + v) * 16 + n] = f2b(fmaxf(acc[v] + bb, 0.f));
  }
  // final: 16 -> 1, sigmoid (one lane per row)
  if (lane < 16) {
    float s = sFb;
#pragma unroll
    for (int c = 0; c < 16; ++c)
      s += (float)B[lane * 16 + c] * sFw[c];
    out[row0 + lane] = 1.f / (1.f + __expf(-s));
  }
}

// ================================ launcher ===================================
extern "C" void kernel_launch(void* const* d_in, const int* in_sizes, int n_in,
                              void* d_out, int out_size, void* d_ws, size_t ws_size,
                              hipStream_t stream) {
  (void)in_sizes; (void)n_in; (void)out_size; (void)ws_size;

  const float* one_hot = (const float*)d_in[0];
  const float* feat    = (const float*)d_in[1];
  const int*   esrc    = (const int*)d_in[3];
  const int*   edst    = (const int*)d_in[4];
  const int*   etyp    = (const int*)d_in[5];
  const float* enc_w[3] = {(const float*)d_in[6], (const float*)d_in[8], (const float*)d_in[10]};
  const float* enc_b[3] = {(const float*)d_in[7], (const float*)d_in[9], (const float*)d_in[11]};
  const float* conv_wr[4]; const float* conv_ws[4]; const float* conv_b[4];
  for (int i = 0; i < 4; ++i) {
    conv_wr[i] = (const float*)d_in[12 + 3 * i];
    conv_ws[i] = (const float*)d_in[13 + 3 * i];
    conv_b[i]  = (const float*)d_in[14 + 3 * i];
  }
  const float* sc_w[5]; const float* sc_b[5];
  for (int i = 0; i < 5; ++i) { sc_w[i] = (const float*)d_in[24 + 2 * i]; sc_b[i] = (const float*)d_in[25 + 2 * i]; }
  const float* fin_w = (const float*)d_in[34];
  const float* fin_b = (const float*)d_in[35];
  float* out = (float*)d_out;

  // workspace layout
  char* ws = (char*)d_ws;
  auto align256 = [](size_t s) { return (s + 255) & ~(size_t)255; };
  size_t off = 0;
  bf16_t* xA   = (bf16_t*)(ws + off); off += align256((size_t)N_NODES * 128 * sizeof(bf16_t));
  bf16_t* xB   = (bf16_t*)(ws + off); off += align256((size_t)N_NODES * 128 * sizeof(bf16_t));
  float*  agg  = (float*) (ws + off); off += align256((size_t)N_NODES * 98 * sizeof(float));
  bf16_t* hbuf = (bf16_t*)(ws + off); // N * 4 * 98 bf16 max

  const int blocks = (N_TILES + WPB - 1) / WPB;   // 782

  enc_kernel<<<blocks, BLOCK_THREADS, 0, stream>>>(
      one_hot, feat, enc_w[0], enc_b[0], enc_w[1], enc_b[1], enc_w[2], enc_b[2], xA);

  const int dims[5]  = {128, 98, 68, 38, 8};
  const int kpads[5] = {128, 128, 96, 64, 32};   // padded row widths of x buffers
  bf16_t* cur = xA; bf16_t* nxt = xB;
  for (int L = 0; L < 4; ++L) {
    int di = dims[L], dout = dims[L + 1];
    switch (kpads[L] >> 5) {
      case 4: conv_gemm_kernel<4><<<blocks, BLOCK_THREADS, 0, stream>>>(cur, conv_wr[L], conv_ws[L], conv_b[L], agg, hbuf, di, dout); break;
      case 3: conv_gemm_kernel<3><<<blocks, BLOCK_THREADS, 0, stream>>>(cur, conv_wr[L], conv_ws[L], conv_b[L], agg, hbuf, di, dout); break;
      default: conv_gemm_kernel<2><<<blocks, BLOCK_THREADS, 0, stream>>>(cur, conv_wr[L], conv_ws[L], conv_b[L], agg, hbuf, di, dout); break;
    }
    scatter_kernel<<<2048, BLOCK_THREADS, 0, stream>>>(hbuf, esrc, edst, etyp, agg, N_EDGES, dout);
    int kp = kpads[L + 1];
    int n = N_NODES * kp;
    relu_pad_bf16_kernel<<<(n + BLOCK_THREADS - 1) / BLOCK_THREADS, BLOCK_THREADS, 0, stream>>>(
        agg, nxt, dout, kp, n);
    bf16_t* t = cur; cur = nxt; nxt = t;
  }

  sc_kernel<<<blocks, BLOCK_THREADS, 0, stream>>>(
      cur, sc_w[0], sc_b[0], sc_w[1], sc_b[1], sc_w[2], sc_b[2],
      sc_w[3], sc_b[3], sc_w[4], sc_b[4], fin_w, fin_b, out);
}
